// MyGAT_6828998000903
// MI455X (gfx1250) — matrix operand order repaired and checked by reading, not compiled, not run on hardware
//
#include <hip/hip_runtime.h>

#define NNODES 50000
#define NEDGES 400000
#define ETOT   (NEDGES + NNODES)
#define NEG_SLOPE 0.2f
#define BN_EPS 1e-5f

typedef __attribute__((ext_vector_type(16))) _Float16 v16h;
typedef __attribute__((ext_vector_type(8)))  float    v8f;

union AFrag { v16h v; unsigned int u[8]; };
union CFrag { v8f  v; float f[8]; };
union H8    { uint4 q; _Float16 h[8]; };
union H2    { unsigned int u; _Float16 h[2]; };

// ---------------------------------------------------------------- utilities
__global__ __launch_bounds__(256) void k_w2h(const float* __restrict__ W,
                                             _Float16* __restrict__ Wh, int n) {
    int i = blockIdx.x * 256 + threadIdx.x;
    if (i < n) Wh[i] = (_Float16)W[i];
}

__global__ __launch_bounds__(256) void k_zero(float* __restrict__ p, long n) {
    long i = (long)blockIdx.x * 256 + threadIdx.x;
    if (i < n) p[i] = 0.f;
}

__global__ __launch_bounds__(256) void k_init_softmax(float* __restrict__ amax,
                                                      float* __restrict__ den, int n) {
    int i = blockIdx.x * 256 + threadIdx.x;
    if (i < n) { amax[i] = -INFINITY; den[i] = 0.f; }
}

__device__ __forceinline__ void atomicMaxFloat(float* addr, float v) {
    // IEEE-754 monotonic bit trick (init value is -inf)
    if (v >= 0.f) atomicMax((int*)addr, __float_as_int(v));
    else          atomicMin((unsigned int*)addr, __float_as_uint(v));
}

// ---------------------------------------------------------------- WMMA GEMM
// H[NNODES,COUT] = X[NNODES,KDIM] * Wh[KDIM,COUT]
// COUT>=64 : 32x64 block tile (2x4 wave tiles).  COUT==16 : 128x16 (8x1).
template <int KDIM, int COUT>
__global__ __launch_bounds__(256)
void k_gemm_wmma(const float* __restrict__ X, const _Float16* __restrict__ Wh,
                 float* __restrict__ H) {
    constexpr int BN    = (COUT >= 64) ? 64 : COUT;   // 64 or 16
    constexpr int WCOLS = BN / 16;                    // 4 or 1
    constexpr int WROWS = 8 / WCOLS;                  // 2 or 8
    constexpr int BM    = WROWS * 16;                 // 32 or 128
    static_assert(COUT % BN == 0, "grid must tile COUT exactly");

    __shared__ _Float16 As[BM][36];   // [row][k], padded
    __shared__ _Float16 BsT[BN][36];  // [col][k] (transposed W chunk)

    const int tid  = threadIdx.x;
    const int lane = tid & 31;
    const int wave = tid >> 5;                 // 0..7
    const int wrow = wave / WCOLS;
    const int wcol = wave % WCOLS;
    const int m    = lane & 15;
    const int half = lane >> 4;
    const int rowBase = blockIdx.x * BM;
    const int colBase = blockIdx.y * BN;

    const int ldrow = tid >> 3;                // 0..31
    const int ldcg  = tid & 7;                 // 0..7

    CFrag acc;
#pragma unroll
    for (int i = 0; i < 8; ++i) acc.f[i] = 0.f;

    for (int k0 = 0; k0 < KDIM; k0 += 32) {
        // --- stage A (fp32 -> f16): BM rows x 32 k
#pragma unroll
        for (int j = 0; j < BM / 32; ++j) {
            int r  = ldrow + j * 32;
            int gr = rowBase + r;
            float4 xv = make_float4(0.f, 0.f, 0.f, 0.f);
            if (gr < NNODES)
                xv = *(const float4*)&X[(size_t)gr * KDIM + k0 + ldcg * 4];
            As[r][ldcg * 4 + 0] = (_Float16)xv.x;
            As[r][ldcg * 4 + 1] = (_Float16)xv.y;
            As[r][ldcg * 4 + 2] = (_Float16)xv.z;
            As[r][ldcg * 4 + 3] = (_Float16)xv.w;
        }
        // --- stage B transposed: 32 k x BN cols (guard-free, exact vectors)
        {
            int kr = k0 + ldrow;               // KDIM multiple of 32
            if constexpr (BN == 64) {
                int cb = colBase + ldcg * 8;
                H8 bv;
                bv.q = *(const uint4*)&Wh[(size_t)kr * COUT + cb];
#pragma unroll
                for (int j = 0; j < 8; ++j)
                    BsT[ldcg * 8 + j][ldrow] = bv.h[j];
            } else {                            // BN == 16
                int cb = colBase + ldcg * 2;
                H2 bv;
                bv.u = *(const unsigned int*)&Wh[(size_t)kr * COUT + cb];
                BsT[ldcg * 2 + 0][ldrow] = bv.h[0];
                BsT[ldcg * 2 + 1][ldrow] = bv.h[1];
            }
        }
        __syncthreads();

        // --- assemble fragments per ISA wave32 layouts
        AFrag a, b;
#pragma unroll
        for (int i = 0; i < 8; ++i) {
            int kbA = ((i < 4) ? (2 * i) : (16 + 2 * (i - 4))) + 8 * half;
            a.u[i] = *(const unsigned int*)&As[wrow * 16 + m][kbA];
            int kbB = ((i < 4) ? (2 * i) : (16 + 2 * (i - 4))) + 8 * half;
            b.u[i] = *(const unsigned int*)&BsT[wcol * 16 + m][kbB];
        }
        acc.v = __builtin_amdgcn_wmma_f32_16x16x32_f16(
            false, a.v, false, b.v, (short)0, acc.v, false, false);
        __syncthreads();
    }

    // --- store C: VGPR i -> M = i + 8*half, N = lane%16 (col always in range)
#pragma unroll
    for (int i = 0; i < 8; ++i) {
        int r = rowBase + wrow * 16 + half * 8 + i;
        int c = colBase + wcol * 16 + m;
        if (r < NNODES) H[(size_t)r * COUT + c] = acc.f[i];
    }
}

// ---------------------------------------------------------------- attention
template <int HEADS, int CH>
__global__ __launch_bounds__(256)
void k_esed(const float* __restrict__ H, const float* __restrict__ asrc,
            const float* __restrict__ adst, float* __restrict__ ES,
            float* __restrict__ ED) {
    constexpr int HC = HEADS * CH;
    int i = blockIdx.x * 256 + threadIdx.x;
    if (i >= NNODES * HEADS) return;
    int node = i / HEADS, h = i % HEADS;
    const float* hp = &H[(size_t)node * HC + h * CH];
    const float* as = &asrc[h * CH];
    const float* ad = &adst[h * CH];
    float es = 0.f, ed = 0.f;
#pragma unroll
    for (int c = 0; c < CH; ++c) { float v = hp[c]; es += v * as[c]; ed += v * ad[c]; }
    ES[i] = es; ED[i] = ed;
}

template <int HEADS>
__global__ __launch_bounds__(256)
void k_edge_max(const int* __restrict__ src, const int* __restrict__ dst,
                const float* __restrict__ ES, const float* __restrict__ ED,
                float* __restrict__ AMAX) {
    long i = (long)blockIdx.x * 256 + threadIdx.x;
    if (i >= (long)ETOT * HEADS) return;
    int e = (int)(i / HEADS), h = (int)(i % HEADS);
    int s, d;
    if (e < NEDGES) { s = src[e]; d = dst[e]; } else { s = d = e - NEDGES; }
    float a = ES[s * HEADS + h] + ED[d * HEADS + h];
    a = a > 0.f ? a : NEG_SLOPE * a;
    atomicMaxFloat(&AMAX[d * HEADS + h], a);
}

template <int HEADS>
__global__ __launch_bounds__(256)
void k_edge_expsum(const int* __restrict__ src, const int* __restrict__ dst,
                   const float* __restrict__ ES, const float* __restrict__ ED,
                   const float* __restrict__ AMAX, float* __restrict__ DEN,
                   float* __restrict__ EXB) {
    long i = (long)blockIdx.x * 256 + threadIdx.x;
    if (i >= (long)ETOT * HEADS) return;
    int e = (int)(i / HEADS), h = (int)(i % HEADS);
    int s, d;
    if (e < NEDGES) { s = src[e]; d = dst[e]; } else { s = d = e - NEDGES; }
    float a = ES[s * HEADS + h] + ED[d * HEADS + h];
    a = a > 0.f ? a : NEG_SLOPE * a;
    float ex = __expf(a - AMAX[d * HEADS + h]);
    EXB[i] = ex;
    atomicAdd(&DEN[d * HEADS + h], ex);
}

template <int HEADS>
__global__ __launch_bounds__(256)
void k_edge_attn(const int* __restrict__ dst, const float* __restrict__ DEN,
                 float* __restrict__ EXB) {
    long i = (long)blockIdx.x * 256 + threadIdx.x;
    if (i >= (long)ETOT * HEADS) return;
    int e = (int)(i / HEADS), h = (int)(i % HEADS);
    int d = (e < NEDGES) ? dst[e] : e - NEDGES;
    EXB[i] = EXB[i] / (DEN[d * HEADS + h] + 1e-16f);
}

template <int HEADS, int CH>
__global__ __launch_bounds__(256)
void k_edge_agg(const int* __restrict__ src, const int* __restrict__ dst,
                const float* __restrict__ EXB, const float* __restrict__ H,
                float* __restrict__ ACC) {
    constexpr int HC = HEADS * CH;
    long i = (long)blockIdx.x * 256 + threadIdx.x;
    if (i >= (long)ETOT * HC) return;
    int e  = (int)(i / HC);
    int cc = (int)(i % HC);
    int h  = cc / CH;
    int s, d;
    if (e < NEDGES) { s = src[e]; d = dst[e]; } else { s = d = e - NEDGES; }
    float w = EXB[(long)e * HEADS + h];
    atomicAdd(&ACC[(size_t)d * HC + cc], H[(size_t)s * HC + cc] * w);
}

// ---------------------------------------------------------------- epilogue
template <int HEADS, int CH, bool CONCAT, bool ELU>
__global__ __launch_bounds__(256)
void k_finalize(const float* __restrict__ ACC, const float* __restrict__ bias,
                const float* __restrict__ g, const float* __restrict__ be,
                const float* __restrict__ rm, const float* __restrict__ rv,
                float* __restrict__ out) {
    constexpr int HC   = HEADS * CH;
    constexpr int DOUT = CONCAT ? HC : CH;
    long i = (long)blockIdx.x * 256 + threadIdx.x;
    if (i >= (long)NNODES * DOUT) return;
    int node = (int)(i / DOUT), j = (int)(i % DOUT);
    float v;
    if constexpr (CONCAT) {
        v = ACC[(size_t)node * HC + j];
    } else {
        float s = 0.f;
#pragma unroll
        for (int h = 0; h < HEADS; ++h) s += ACC[(size_t)node * HC + h * CH + j];
        v = s * (1.f / (float)HEADS);
    }
    v += bias[j];
    if constexpr (ELU) v = v > 0.f ? v : (__expf(v) - 1.f);
    v = (v - rm[j]) * rsqrtf(rv[j] + BN_EPS) * g[j] + be[j];
    out[(size_t)node * DOUT + j] = v;
}

// ---------------------------------------------------------------- per-layer driver
struct Ws {
    float* XBUF; float* HBUF; float* ACC; _Float16* WH;
    float* ES; float* ED; float* AMAX; float* DEN; float* EXB;
};

static inline unsigned int cdiv(long a, long b) { return (unsigned int)((a + b - 1) / b); }

template <int DIN, int HEADS, int CH, bool CONCAT, bool ELU>
static void run_layer(const float* Xin, void* const* d_in, int l, const int* src,
                      const int* dst, float* outp, const Ws& ws, hipStream_t stream) {
    constexpr int HC = HEADS * CH;
    const float* Wl = (const float*)d_in[3 + 8 * l];
    const float* as = (const float*)d_in[4 + 8 * l];
    const float* ad = (const float*)d_in[5 + 8 * l];
    const float* bs = (const float*)d_in[6 + 8 * l];
    const float* gg = (const float*)d_in[7 + 8 * l];
    const float* bb = (const float*)d_in[8 + 8 * l];
    const float* rm = (const float*)d_in[9 + 8 * l];
    const float* rv = (const float*)d_in[10 + 8 * l];

    // 1) W -> f16
    k_w2h<<<cdiv(DIN * HC, 256), 256, 0, stream>>>(Wl, ws.WH, DIN * HC);

    // 2) H = X * W (WMMA, fully specialized tiles)
    constexpr int BN = (HC >= 64) ? 64 : HC;
    constexpr int BM = (8 / (BN / 16)) * 16;
    dim3 ggrid(cdiv(NNODES, BM), HC / BN);
    k_gemm_wmma<DIN, HC><<<ggrid, 256, 0, stream>>>(Xin, ws.WH, ws.HBUF);

    // 3) per-node logits
    k_esed<HEADS, CH><<<cdiv((long)NNODES * HEADS, 256), 256, 0, stream>>>(
        ws.HBUF, as, ad, ws.ES, ws.ED);

    // 4) softmax state + ACC init
    k_init_softmax<<<cdiv((long)NNODES * HEADS, 256), 256, 0, stream>>>(
        ws.AMAX, ws.DEN, NNODES * HEADS);
    k_zero<<<cdiv((long)NNODES * HC, 256), 256, 0, stream>>>(ws.ACC, (long)NNODES * HC);

    // 5) segment max / exp-sum / normalize
    long eh = (long)ETOT * HEADS;
    k_edge_max<HEADS><<<cdiv(eh, 256), 256, 0, stream>>>(src, dst, ws.ES, ws.ED, ws.AMAX);
    k_edge_expsum<HEADS><<<cdiv(eh, 256), 256, 0, stream>>>(src, dst, ws.ES, ws.ED,
                                                            ws.AMAX, ws.DEN, ws.EXB);
    k_edge_attn<HEADS><<<cdiv(eh, 256), 256, 0, stream>>>(dst, ws.DEN, ws.EXB);

    // 6) weighted scatter aggregation
    k_edge_agg<HEADS, CH><<<cdiv((long)ETOT * HC, 256), 256, 0, stream>>>(
        src, dst, ws.EXB, ws.HBUF, ws.ACC);

    // 7) bias + (ELU) + BN
    constexpr int DOUT = CONCAT ? HC : CH;
    k_finalize<HEADS, CH, CONCAT, ELU><<<cdiv((long)NNODES * DOUT, 256), 256, 0, stream>>>(
        ws.ACC, bs, gg, bb, rm, rv, outp);
}

// ---------------------------------------------------------------- launcher
extern "C" void kernel_launch(void* const* d_in, const int* in_sizes, int n_in,
                              void* d_out, int out_size, void* d_ws, size_t ws_size,
                              hipStream_t stream) {
    const float* x   = (const float*)d_in[0];
    const int*   src = (const int*)d_in[1];
    const int*   dst = (const int*)d_in[2];

    char* w = (char*)d_ws;
    auto alloc = [&](size_t bytes) -> void* {
        void* p = (void*)w; w += (bytes + 255) & ~(size_t)255; return p;
    };
    Ws ws;
    ws.XBUF = (float*)alloc((size_t)NNODES * 256 * 4);
    ws.HBUF = (float*)alloc((size_t)NNODES * 256 * 4);
    ws.ACC  = (float*)alloc((size_t)NNODES * 256 * 4);
    ws.WH   = (_Float16*)alloc((size_t)256 * 256 * 2);
    ws.ES   = (float*)alloc((size_t)NNODES * 8 * 4);
    ws.ED   = (float*)alloc((size_t)NNODES * 8 * 4);
    ws.AMAX = (float*)alloc((size_t)NNODES * 8 * 4);
    ws.DEN  = (float*)alloc((size_t)NNODES * 8 * 4);
    ws.EXB  = (float*)alloc((size_t)ETOT * 8 * 4);

    run_layer<128, 8, 32, true,  true >(x,       d_in, 0, src, dst, ws.XBUF, ws, stream);
    run_layer<256, 8, 16, true,  true >(ws.XBUF, d_in, 1, src, dst, ws.XBUF, ws, stream);
    run_layer<128, 8, 8,  true,  true >(ws.XBUF, d_in, 2, src, dst, ws.XBUF, ws, stream);
    run_layer<64,  1, 16, false, false>(ws.XBUF, d_in, 3, src, dst, (float*)d_out, ws, stream);
}
